// SpatioTemporalMixHopLayer_31842887532842
// MI455X (gfx1250) — matrix operand order repaired and checked by reading, not compiled, not run on hardware
//
#include <hip/hip_runtime.h>

// ---------------------------------------------------------------------------
// SpatioTemporalMixHop for MI455X (gfx1250), wave32 + WMMA bf16 + TDM/async.
// B=8, C=32, N=256, T=64, O=256, branches: adj^{0,1,2}, Td^{0,1}.
// ---------------------------------------------------------------------------

#define DEV static __device__ __forceinline__

typedef __attribute__((ext_vector_type(16))) __bf16 v16bf;
typedef __attribute__((ext_vector_type(8)))  __bf16 v8bf;
typedef __attribute__((ext_vector_type(8)))  float  v8f;
typedef __attribute__((ext_vector_type(4)))  float  v4f;
typedef __attribute__((ext_vector_type(4)))  int    v4i;
typedef __attribute__((ext_vector_type(8)))  int    v8i_;
typedef __attribute__((ext_vector_type(4)))  unsigned int v4u;

constexpr int Bn = 8, Cc = 32, Nn = 256, Tt = 64, Oo = 256;
constexpr int NT   = Nn * Tt;        // 16384 rows per batch
constexpr int NBR  = 5;
constexpr int OUTC = NBR * Oo;       // 1280 output channels
constexpr float SLOPE = 0.01f;

DEV v16bf cat16(v8bf lo, v8bf hi) {
  return __builtin_shufflevector(lo, hi, 0,1,2,3,4,5,6,7,8,9,10,11,12,13,14,15);
}

DEV v8f wmma_bf16(v16bf a, v16bf b, v8f c) {
  // D = A(16x32) * B(32x16) + C, f32 accumulate
  return __builtin_amdgcn_wmma_f32_16x16x32_bf16(
      /*neg_a=*/false, a, /*neg_b=*/false, b,
      /*c_mod=*/(short)0, c, /*reuse_a=*/false, /*reuse_b=*/false);
}

DEV float leaky(float v) { return v > 0.f ? v : v * SLOPE; }

// ---- CDNA5 async-LDS / Tensor-Data-Mover staging (guarded, with fallback) --
#define AS1 __attribute__((address_space(1)))
#define AS3 __attribute__((address_space(3)))

#if __has_builtin(__builtin_amdgcn_global_load_async_to_lds_b128)
#define HAS_ASYNC_LDS 1
DEV void async_cp16(void* lds_dst, const void* gsrc) {
  // 4-arg form, param0 = global int32x4*, param1 = LDS int32x4*
  __builtin_amdgcn_global_load_async_to_lds_b128(
      (AS1 v4i*)gsrc, (AS3 v4i*)lds_dst, /*offset=*/0, /*cpol=*/0);
}
DEV void async_wait0() {
#if __has_builtin(__builtin_amdgcn_s_wait_asynccnt)
  __builtin_amdgcn_s_wait_asynccnt(0);
#else
  asm volatile("s_wait_asynccnt 0x0" ::: "memory");
#endif
}
#endif

#if __has_builtin(__builtin_amdgcn_tensor_load_to_lds)
#define HAS_TDM 1
DEV void tensor_wait0() {
#if __has_builtin(__builtin_amdgcn_s_wait_tensorcnt)
  __builtin_amdgcn_s_wait_tensorcnt(0);
#else
  asm volatile("s_wait_tensorcnt 0x0" ::: "memory");
#endif
}
// DMA a contiguous (8192 x rows) bf16 tile from global to LDS via the TDM.
// D# per CDNA5 ISA ch.8: group0 = {flags, lds_addr, gaddr_lo, gaddr_hi|type},
// group1 = dims/strides (2D), groups 2/3 + trailing group zero (<=2D tensor).
DEV void tdm_load_contig_bf16(void* lds_dst, const void* gsrc, int rows) {
  unsigned lo = (unsigned)(size_t)(AS3 void*)lds_dst;       // LDS byte offset
  unsigned long long ga = (unsigned long long)(size_t)gsrc; // 57-bit global
  v4u g0 = { 1u,                                            // count=1, user D#
             lo,                                            // lds_addr
             (unsigned)ga,                                  // global_addr[31:0]
             (unsigned)((ga >> 32) & 0x01FFFFFFu) | (2u << 30) }; // [56:32]|type=2
  v8i_ g1 = { (int)(1u << 16),        // wg_mask=0, data_size=1 (2 bytes)
              (int)(0x2000u << 16),   // tensor_dim0 = 8192 (low16 @ [31:16])
              (int)((unsigned)rows << 16), // tdim0 hi=0 | tensor_dim1 = rows
              (int)(0x2000u << 16),   // tdim1 hi=0 | tile_dim0 = 8192
              rows,                   // tile_dim1 = rows | tile_dim2 = 0
              8192,                   // tensor_dim0_stride[31:0] = 8192
              0, 0 };                 // stride hi / tensor_dim1_stride = 0
  v4i gz4 = {0, 0, 0, 0};
  v8i_ gz8 = {0, 0, 0, 0, 0, 0, 0, 0};
  __builtin_amdgcn_tensor_load_to_lds(g0, g1, gz4, gz4, gz8, /*cpol=*/0);
}
#endif

// ---------------------------------------------------------------------------
// Prep 1: x[b][c][nt] (f32) -> xT[b][nt][c] (bf16), LDS tiled transpose.
// grid = Bn * (NT/32), block = 256
// ---------------------------------------------------------------------------
__global__ void __launch_bounds__(256)
prep_x(const float* __restrict__ x, __bf16* __restrict__ xT) {
  __shared__ float tile[32][33];
  int b   = blockIdx.x >> 9;              // NT/32 = 512 tiles per batch
  int nt0 = (blockIdx.x & 511) * 32;
  const float* xb = x + (size_t)b * Cc * NT;
  for (int i = threadIdx.x; i < 32 * 32; i += 256) {
    int c = i >> 5, j = i & 31;           // j over nt -> coalesced read
    tile[c][j] = xb[(size_t)c * NT + nt0 + j];
  }
  __syncthreads();
  __bf16* dst = xT + ((size_t)b * NT + nt0) * Cc;
  for (int i = threadIdx.x; i < 32 * 32; i += 256) {
    int r = i >> 5, c = i & 31;           // c fastest -> coalesced write
    dst[r * Cc + c] = (__bf16)tile[c][r];
  }
}

// ---------------------------------------------------------------------------
// Prep 2: adj -> bf16; W_a/W_t -> Wt5[k][o][c] bf16 (k=0..2 Wa, 3..4 Wt);
//         biases -> bias5[k][o] f32.   grid = 2048, block = 256
// ---------------------------------------------------------------------------
__global__ void __launch_bounds__(256)
prep_misc(const float* __restrict__ adj, const float* __restrict__ W_a,
          const float* __restrict__ b_a, const float* __restrict__ W_t,
          const float* __restrict__ b_t, __bf16* __restrict__ adjb,
          __bf16* __restrict__ Wt5, float* __restrict__ bias5) {
  int i = blockIdx.x * 256 + threadIdx.x;
  if (i < Bn * Nn * Nn) adjb[i] = (__bf16)adj[i];
  if (i < NBR * Oo * Cc) {
    int c = i & 31, o = (i >> 5) & 255, k = i >> 13;
    float w = (k < 3) ? W_a[(k * Cc + c) * Oo + o]
                      : W_t[((k - 3) * Cc + c) * Oo + o];
    Wt5[i] = (__bf16)w;
  }
  if (i < NBR * Oo) bias5[i] = (i < 3 * Oo) ? b_a[i] : b_t[i - 3 * Oo];
}

// ---------------------------------------------------------------------------
// Projection GEMM, t-contiguous strips: branches {0,3,4}.
//   branch 0 -> leaky -> out[:, 0:256];  branch 3 -> leaky -> out[:, 768:1024]
//   branch 4 -> h4[b][o][n][t] bf16 (for temporal diff)
// grid = Bn * (NT/128) = 1024, block = 256 (8 waves, wave w -> 16-row tile w)
// ---------------------------------------------------------------------------
__global__ void __launch_bounds__(256)
gemm_proj_t(const __bf16* __restrict__ xT, const __bf16* __restrict__ Wt5,
            const float* __restrict__ bias5, float* __restrict__ out,
            __bf16* __restrict__ h4) {
  __shared__ __attribute__((aligned(32))) __bf16 ldsW[3 * Oo * Cc]; // 48 KB
  int b   = blockIdx.x >> 7;
  int nt0 = (blockIdx.x & 127) * 128;
  // stage W for branches 0,3,4 (compacted kb = 0,1,2)
  for (int i = threadIdx.x; i < 3 * Oo * Cc / 8; i += 256) {
    int idx = i * 8;
    int kk  = idx >> 13;                      // 0..2
    int br  = (kk == 0) ? 0 : (kk + 2);       // 0,3,4
    int rem = idx & (Oo * Cc - 1);
#if defined(HAS_ASYNC_LDS)
    async_cp16(&ldsW[idx], &Wt5[(size_t)br * Oo * Cc + rem]);
#else
    *(v8bf*)&ldsW[idx] = *(const v8bf*)&Wt5[(size_t)br * Oo * Cc + rem];
#endif
  }
#if defined(HAS_ASYNC_LDS)
  async_wait0();
#endif
  __syncthreads();

  int wave = threadIdx.x >> 5, lane = threadIdx.x & 31;
  int l15  = lane & 15;
  int row  = nt0 + wave * 16 + l15;           // nt row; 16 rows = one n, 16 t's
  int koff = (lane < 16) ? 0 : 16;            // A frag K-run select
  int woff = (lane < 16) ? 0 : 16;            // B frag K-half select
  int m0   = (lane < 16) ? 0 : 8;             // D rows held by this lane

  const __bf16* arow = xT + ((size_t)b * NT + row) * Cc;
  v16bf afrag = cat16(*(const v8bf*)(arow + koff),
                      *(const v8bf*)(arow + koff + 8));

  for (int kb = 0; kb < 3; ++kb) {
    int br = (kb == 0) ? 0 : (kb + 2);
    for (int ot = 0; ot < 16; ++ot) {
      int o = ot * 16 + l15;
      v16bf bfrag = *(const v16bf*)&ldsW[(kb * Oo + o) * Cc + woff];
      v8f c = {0.f,0.f,0.f,0.f,0.f,0.f,0.f,0.f};
      c = wmma_bf16(afrag, bfrag, c);
      float bias = bias5[br * Oo + o];
      if (br == 4) {                          // h4: bf16, t contiguous
        v8bf pk;
#pragma unroll
        for (int r = 0; r < 8; ++r) pk[r] = (__bf16)(c[r] + bias);
        *(v8bf*)(h4 + ((size_t)b * Oo + o) * NT + nt0 + wave * 16 + m0) = pk;
      } else {                                // direct output, leaky
        int ch = (br == 0) ? o : (3 * Oo + o);
        float* dst = out + ((size_t)b * OUTC + ch) * NT + nt0 + wave * 16 + m0;
        v4f lo, hi;
#pragma unroll
        for (int r = 0; r < 4; ++r) { lo[r] = leaky(c[r] + bias); hi[r] = leaky(c[r + 4] + bias); }
        *(v4f*)dst = lo;  *(v4f*)(dst + 4) = hi;
      }
    }
  }
}

// ---------------------------------------------------------------------------
// Projection GEMM, n-contiguous strips: branches {1,2}.
//   h1, h2 stored as [b][o][t][n] bf16 (n contiguous -> diffusion B-fragments
//   and intermediate stores are contiguous 16B per lane).
// grid = Bn * Tt = 512, block = 256 (8 waves; wave w -> n-tiles 2w, 2w+1)
// ---------------------------------------------------------------------------
__global__ void __launch_bounds__(256)
gemm_proj_n(const __bf16* __restrict__ xT, const __bf16* __restrict__ Wt5,
            const float* __restrict__ bias5, __bf16* __restrict__ h1,
            __bf16* __restrict__ h2) {
  __shared__ __attribute__((aligned(32))) __bf16 ldsW[2 * Oo * Cc]; // 32 KB
  int b = blockIdx.x >> 6;
  int t = blockIdx.x & 63;
  for (int i = threadIdx.x * 8; i < 2 * Oo * Cc; i += 256 * 8) {
#if defined(HAS_ASYNC_LDS)
    async_cp16(&ldsW[i], &Wt5[(size_t)1 * Oo * Cc + i]);   // k=1,2 contiguous
#else
    *(v8bf*)&ldsW[i] = *(const v8bf*)&Wt5[(size_t)1 * Oo * Cc + i];
#endif
  }
#if defined(HAS_ASYNC_LDS)
  async_wait0();
#endif
  __syncthreads();

  int wave = threadIdx.x >> 5, lane = threadIdx.x & 31;
  int l15  = lane & 15;
  int koff = (lane < 16) ? 0 : 16;
  int woff = (lane < 16) ? 0 : 16;
  int m0   = (lane < 16) ? 0 : 8;
  __bf16* houts[2] = {h1, h2};

#pragma unroll
  for (int sub = 0; sub < 2; ++sub) {
    int n0 = (wave * 2 + sub) * 16;
    int nt = (n0 + l15) * Tt + t;             // rows = 16 consecutive n, fixed t
    const __bf16* arow = xT + ((size_t)b * NT + nt) * Cc;
    v16bf afrag = cat16(*(const v8bf*)(arow + koff),
                        *(const v8bf*)(arow + koff + 8));
    for (int kb = 0; kb < 2; ++kb) {
      for (int ot = 0; ot < 16; ++ot) {
        int o = ot * 16 + l15;
        v16bf bfrag = *(const v16bf*)&ldsW[(kb * Oo + o) * Cc + woff];
        v8f c = {0.f,0.f,0.f,0.f,0.f,0.f,0.f,0.f};
        c = wmma_bf16(afrag, bfrag, c);
        float bias = bias5[(1 + kb) * Oo + o];
        v8bf pk;
#pragma unroll
        for (int r = 0; r < 8; ++r) pk[r] = (__bf16)(c[r] + bias);
        // h[b][o][t][n]: lane stores 8 consecutive n (16B contiguous)
        *(v8bf*)(houts[kb] + (((size_t)b * Oo + o) * Tt + t) * Nn + n0 + m0) = pk;
      }
    }
  }
}

// ---------------------------------------------------------------------------
// Diffusion: out(n,t) = adj[b](n,m) @ h(m,t) per (b,o).
//   adj[b] (256x256 bf16, 128 KB) staged in LDS once per block — via the
//   Tensor Data Mover (one tensor_load_to_lds DMA) when available, else
//   async global->LDS b128 copies, else plain VGPR round-trip.
// grid = Bn * (Oo/4) = 512, block = 256, dynamic smem = 128KB + 8KB
// ---------------------------------------------------------------------------
__global__ void __launch_bounds__(256)
diffuse(const __bf16* __restrict__ adjb, const __bf16* __restrict__ hin,
        float* __restrict__ out_f32, __bf16* __restrict__ out_bf,
        int ch_base, int final_mode) {
  extern __shared__ char smem[];
  __bf16* ldsA = (__bf16*)smem;                       // 256*256 bf16 = 128 KB
  float*  ldsT = (float*)(smem + Nn * Nn * 2);        // 8 waves * 256 f32 = 8 KB

  int b  = blockIdx.x >> 6;
  int o0 = (blockIdx.x & 63) * 4;

  // ---- stage adjacency for this batch --------------------------------------
#if defined(HAS_TDM)
  if ((threadIdx.x >> 5) == 0) {       // wave 0 issues one TDM DMA (128 KB)
    tdm_load_contig_bf16(ldsA, adjb + (size_t)b * Nn * Nn, /*rows=*/8);
    tensor_wait0();                    // TENSORcnt is per-wave
  }
  __syncthreads();
#elif defined(HAS_ASYNC_LDS)
  {
    const char* src = (const char*)(adjb + (size_t)b * Nn * Nn);
    char* dst = (char*)ldsA;
    for (int i = threadIdx.x; i < Nn * Nn * 2 / 16; i += 256)
      async_cp16(dst + i * 16, src + i * 16);
    async_wait0();
  }
  __syncthreads();
#else
  {
    const v8bf* src = (const v8bf*)(adjb + (size_t)b * Nn * Nn);
    v8bf* dst = (v8bf*)ldsA;
    for (int i = threadIdx.x; i < Nn * Nn / 8; i += 256) dst[i] = src[i];
  }
  __syncthreads();
#endif

  int wave = threadIdx.x >> 5, lane = threadIdx.x & 31;
  int l15  = lane & 15;
  int tt   = wave & 3;            // t-tile 0..3
  int ng   = wave >> 2;           // n-group 0..1 (tiles ng*8 .. ng*8+7)
  int m0   = (lane < 16) ? 0 : 8;
  int half = (lane < 16) ? 0 : 16;
  int ahlf = (lane < 16) ? 0 : 8;

  for (int oi = 0; oi < 4; ++oi) {
    int o = o0 + oi;
    const __bf16* hrow = hin + (((size_t)b * Oo + o) * Tt + tt * 16 + l15) * Nn;
    if (oi < 3)  // prefetch next o's slab (global_prefetch_b8)
      __builtin_prefetch(hin + (((size_t)b * Oo + o + 1) * Tt + tt * 16 + l15) * Nn, 0, 0);

    v16bf bf[8];                  // B-fragments for the full K=256, one t-tile
#pragma unroll
    for (int kk = 0; kk < 8; ++kk)
      bf[kk] = *(const v16bf*)(hrow + kk * 32 + half);

    for (int ntile = 0; ntile < 8; ++ntile) {
      int n0 = (ng * 8 + ntile) * 16;
      v8f c = {0.f,0.f,0.f,0.f,0.f,0.f,0.f,0.f};
#pragma unroll
      for (int kk = 0; kk < 8; ++kk) {
        const __bf16* ar = ldsA + (n0 + l15) * Nn + kk * 32 + half;
        v16bf af = cat16(*(const v8bf*)ar, *(const v8bf*)(ar + 8));
        c = wmma_bf16(af, bf[kk], c);
      }
      if (final_mode) {
        // leaky + per-wave LDS transpose so stores are 32B-contiguous over t
        float* tl = ldsT + wave * 256;
#pragma unroll
        for (int r = 0; r < 8; ++r) tl[(m0 + r) * 16 + l15] = leaky(c[r]);
        v4f lo = *(const v4f*)(tl + l15 * 16 + ahlf);      // same-wave LDS RAW:
        v4f hi = *(const v4f*)(tl + l15 * 16 + ahlf + 4);  // in-order, dscnt wait
        float* dst = out_f32 + (((size_t)b * OUTC + ch_base + o) * Nn + n0 + l15) * Tt
                     + tt * 16 + ahlf;
        *(v4f*)dst = lo;  *(v4f*)(dst + 4) = hi;
      } else {
        v8bf pk;
#pragma unroll
        for (int r = 0; r < 8; ++r) pk[r] = (__bf16)c[r];
        *(v8bf*)(out_bf + (((size_t)b * Oo + o) * Tt + tt * 16 + l15) * Nn + n0 + m0) = pk;
      }
    }
  }
}

// ---------------------------------------------------------------------------
// Temporal branch d=1: out = leaky(h4[n-1] - h4[n]) -> out[:, 1024:1280]
// grid = B*O*NT/256 = 131072, block = 256
// ---------------------------------------------------------------------------
__global__ void __launch_bounds__(256)
tdiff(const __bf16* __restrict__ h4, float* __restrict__ out) {
  size_t i = (size_t)blockIdx.x * 256 + threadIdx.x;   // over B*O*NT
  int nt = (int)(i & (NT - 1));
  float cur  = (float)h4[i];
  float prev = (nt >= Tt) ? (float)h4[i - Tt] : 0.f;
  float v = leaky(prev - cur);
  size_t bo = i >> 14;                                  // b*O + o
  size_t b = bo >> 8, o = bo & 255;
  out[(((b * OUTC) + 4 * Oo + o) << 14) + nt] = v;
}

// ---------------------------------------------------------------------------
extern "C" void kernel_launch(void* const* d_in, const int* in_sizes, int n_in,
                              void* d_out, int out_size, void* d_ws, size_t ws_size,
                              hipStream_t stream) {
  const float* x   = (const float*)d_in[0];
  const float* adj = (const float*)d_in[1];
  const float* W_a = (const float*)d_in[2];
  const float* b_a = (const float*)d_in[3];
  const float* W_t = (const float*)d_in[4];
  const float* b_t = (const float*)d_in[5];
  float* out = (float*)d_out;

  char* ws = (char*)d_ws;
  auto align256 = [](size_t v) { return (v + 255) & ~(size_t)255; };
  size_t off = 0;
  __bf16* xT    = (__bf16*)(ws + off); off = align256(off + (size_t)Bn * NT * Cc * 2);
  __bf16* adjb  = (__bf16*)(ws + off); off = align256(off + (size_t)Bn * Nn * Nn * 2);
  __bf16* Wt5   = (__bf16*)(ws + off); off = align256(off + (size_t)NBR * Oo * Cc * 2);
  float*  bias5 = (float*) (ws + off); off = align256(off + (size_t)NBR * Oo * 4);
  size_t hbytes = (size_t)Bn * Oo * NT * 2;            // 67 MB each
  __bf16* h1 = (__bf16*)(ws + off); off = align256(off + hbytes);
  __bf16* h2 = (__bf16*)(ws + off); off = align256(off + hbytes);
  __bf16* hx = (__bf16*)(ws + off); off = align256(off + hbytes);  // h4, then g2
  (void)ws_size; (void)in_sizes; (void)n_in; (void)out_size;

  constexpr int DIFF_SMEM = Nn * Nn * 2 + 8 * 256 * 4;  // 136 KB dynamic LDS
  (void)hipFuncSetAttribute((const void*)diffuse,
                            hipFuncAttributeMaxDynamicSharedMemorySize, DIFF_SMEM);

  prep_x   <<<Bn * (NT / 32), 256, 0, stream>>>(x, xT);
  prep_misc<<<2048, 256, 0, stream>>>(adj, W_a, b_a, W_t, b_t, adjb, Wt5, bias5);

  gemm_proj_t<<<Bn * (NT / 128), 256, 0, stream>>>(xT, Wt5, bias5, out, hx);
  gemm_proj_n<<<Bn * Tt,         256, 0, stream>>>(xT, Wt5, bias5, h1, h2);

  tdiff<<<(size_t)Bn * Oo * NT / 256, 256, 0, stream>>>(hx, out);  // consumes h4

  // branch 1: one hop, straight to output (channels 256..511)
  diffuse<<<Bn * (Oo / 4), 256, DIFF_SMEM, stream>>>(adjb, h1, out, nullptr, 1 * Oo, 1);
  // branch 2: hop 1 -> hx (reuses h4 space, already consumed), hop 2 -> output
  diffuse<<<Bn * (Oo / 4), 256, DIFF_SMEM, stream>>>(adjb, h2, nullptr, hx, 0, 0);
  diffuse<<<Bn * (Oo / 4), 256, DIFF_SMEM, stream>>>(adjb, hx, out, nullptr, 2 * Oo, 1);
}
